// ParallelMLP_1185410974369
// MI455X (gfx1250) — matrix operand order repaired and checked by reading, not compiled, not run on hardware
//
#include <hip/hip_runtime.h>
#include <hip/hip_bf16.h>

// ---------------- problem constants ----------------
#define N_TOK    8192
#define D_MODEL  2048
#define HIDDEN   2048
#define NEXP     8
#define TOPK     2
#define CAP      2560                 // int(1.25 * 2 * 8192 / 8)
#define T_ASSIGN (N_TOK * TOPK)      // 16384
#define EROWS    (NEXP * CAP)        // 20480 padded rows

typedef __attribute__((ext_vector_type(16))) __bf16 v16bf;
typedef __attribute__((ext_vector_type(8)))  float  v8f;

union FragAB { v16bf v; uint4 q[2]; };
union Pack8  { __bf16 h[8]; uint4 q; };

// ---------------------------------------------------------------------------
// 1) Routing: histogram + stable counting-sort ranks (single block, 256 thr,
//    each thread owns a contiguous 64-assignment segment -> stability).
// ---------------------------------------------------------------------------
__global__ __launch_bounds__(256) void moe_route(const int* __restrict__ eidx,
                                                 int* __restrict__ counts,
                                                 int* __restrict__ slot,
                                                 int* __restrict__ gtok,
                                                 int* __restrict__ counts_out) {
  __shared__ int segHist[256][NEXP];
  const int tid  = threadIdx.x;
  const int base = tid * (T_ASSIGN / 256);   // 64 per thread

  int lh[NEXP];
#pragma unroll
  for (int e = 0; e < NEXP; ++e) lh[e] = 0;
  for (int j = 0; j < T_ASSIGN / 256; ++j) lh[eidx[base + j]]++;
#pragma unroll
  for (int e = 0; e < NEXP; ++e) segHist[tid][e] = lh[e];
  __syncthreads();

  if (tid < NEXP) {                       // serial exclusive scan per expert
    int run = 0;
    for (int s = 0; s < 256; ++s) {
      int t = segHist[s][tid];
      segHist[s][tid] = run;
      run += t;
    }
    counts[tid]     = run;
    counts_out[tid] = run;                // int32 bits into d_out tail
  }
  __syncthreads();

  int r[NEXP];
#pragma unroll
  for (int e = 0; e < NEXP; ++e) r[e] = 0;
  for (int j = 0; j < T_ASSIGN / 256; ++j) {
    const int t    = base + j;
    const int e    = eidx[t];
    const int rank = segHist[tid][e] + r[e];
    r[e]++;
    if (rank < CAP) {
      const int sl = e * CAP + rank;
      slot[t]  = sl;
      gtok[sl] = t >> 1;                  // token = t / TOPK
    } else {
      slot[t] = -1;                       // capacity-dropped
    }
  }
}

// ---------------------------------------------------------------------------
// 2) fp32 (n x n) -> bf16 transposed (per expert): dst[n][k] = src[k][n]
// ---------------------------------------------------------------------------
__global__ __launch_bounds__(1024) void transpose_cvt(const float* __restrict__ src,
                                                      __bf16* __restrict__ dst) {
  __shared__ float tile[32][33];
  const int n = 2048;
  const size_t eo = (size_t)blockIdx.z * n * n;
  const int x = blockIdx.x * 32 + threadIdx.x;
  const int y = blockIdx.y * 32 + threadIdx.y;
  tile[threadIdx.y][threadIdx.x] = src[eo + (size_t)y * n + x];
  __syncthreads();
  const int xo = blockIdx.y * 32 + threadIdx.x;
  const int yo = blockIdx.x * 32 + threadIdx.y;
  dst[eo + (size_t)yo * n + xo] = (__bf16)tile[threadIdx.x][threadIdx.y];
}

// ---------------------------------------------------------------------------
// 3) Gather tokens into padded per-expert bf16 matrix (E*CAP, D_MODEL)
// ---------------------------------------------------------------------------
__global__ __launch_bounds__(256) void gather_xg(const float* __restrict__ x,
                                                 const int* __restrict__ counts,
                                                 const int* __restrict__ gtok,
                                                 __bf16* __restrict__ xg) {
  const int s = blockIdx.x;              // padded row
  const int e = s / CAP;
  const int i = s - e * CAP;
  const int c = threadIdx.x * 8;
  Pack8 p;
  if (i < counts[e]) {
    const int tok = gtok[s];
    const float4 f0 = *(const float4*)(x + (size_t)tok * D_MODEL + c);
    const float4 f1 = *(const float4*)(x + (size_t)tok * D_MODEL + c + 4);
    p.h[0] = (__bf16)f0.x; p.h[1] = (__bf16)f0.y; p.h[2] = (__bf16)f0.z; p.h[3] = (__bf16)f0.w;
    p.h[4] = (__bf16)f1.x; p.h[5] = (__bf16)f1.y; p.h[6] = (__bf16)f1.z; p.h[7] = (__bf16)f1.w;
  } else {
    p.q = make_uint4(0u, 0u, 0u, 0u);
  }
  *(uint4*)(xg + (size_t)s * D_MODEL + c) = p.q;
}

// ---------------------------------------------------------------------------
// Fragment loaders (documented gfx1250 16-bit WMMA VGPR layouts)
//   A 16x32: lane%16 = M row; 16B at K+8*(lane/16), 16B at +16 more
//   B 32x16 (from transposed weight, K contiguous per N row):
//            lane%16 = N col; 16B at K+8*(lane/16), 16B at +16 more
// ---------------------------------------------------------------------------
__device__ __forceinline__ void load_fragA(FragAB& f, const __bf16* base,
                                           int row, int k0, int half) {
  const __bf16* p = base + (size_t)row * D_MODEL + k0 + half * 8;
  f.q[0] = *(const uint4*)p;
  f.q[1] = *(const uint4*)(p + 16);
}
__device__ __forceinline__ void load_fragB(FragAB& f, const __bf16* baseT,
                                           int nrow, int k0, int half) {
  const __bf16* p = baseT + (size_t)nrow * D_MODEL + k0 + half * 8;
  f.q[0] = *(const uint4*)p;
  f.q[1] = *(const uint4*)(p + 16);
}

// ---------------------------------------------------------------------------
// 4) GEMM1 fused SwiGLU: H = silu(Xg @ W1) * (Xg @ W3)   (bf16 out)
//    256 thr = 8 waves (2 rows x 4 cols); block tile 128x128;
//    wave tile 64x32 per matrix (4 M-frags x 2 N-frags, dual accumulators)
//    => 16 WMMAs per K-step vs 16 b128 loads.
// ---------------------------------------------------------------------------
__global__ __launch_bounds__(256) void gemm1_swiglu(const __bf16* __restrict__ xg,
                                                    const __bf16* __restrict__ w1t,
                                                    const __bf16* __restrict__ w3t,
                                                    __bf16* __restrict__ hout) {
  const int mblk = blockIdx.x;                 // 0..159 over E*CAP/128
  const int e    = mblk / (CAP / 128);
  const int tid  = threadIdx.x;
  const int wave = tid >> 5, lane = tid & 31;
  const int wrow = wave & 1, wcol = wave >> 1;       // 2 x 4 wave grid
  const int half = lane >> 4, l16 = lane & 15;
  const int m0 = mblk * 128 + wrow * 64;             // 64-row wave tile
  const int n0 = blockIdx.y * 128 + wcol * 32;       // 32-col wave tile
  const __bf16* W1 = w1t + (size_t)e * HIDDEN * D_MODEL;
  const __bf16* W3 = w3t + (size_t)e * HIDDEN * D_MODEL;

  v8f acc1[4][2], acc3[4][2];
#pragma unroll
  for (int mi = 0; mi < 4; ++mi)
#pragma unroll
    for (int nj = 0; nj < 2; ++nj) {
      acc1[mi][nj] = (v8f)(0.f);
      acc3[mi][nj] = (v8f)(0.f);
    }

  for (int k0 = 0; k0 < D_MODEL; k0 += 32) {
    FragAB a[4], b1[2], b3[2];
#pragma unroll
    for (int mi = 0; mi < 4; ++mi)
      load_fragA(a[mi], xg, m0 + mi * 16 + l16, k0, half);
#pragma unroll
    for (int nj = 0; nj < 2; ++nj) {
      load_fragB(b1[nj], W1, n0 + nj * 16 + l16, k0, half);
      load_fragB(b3[nj], W3, n0 + nj * 16 + l16, k0, half);
    }
#pragma unroll
    for (int mi = 0; mi < 4; ++mi)
#pragma unroll
      for (int nj = 0; nj < 2; ++nj) {
        acc1[mi][nj] = __builtin_amdgcn_wmma_f32_16x16x32_bf16(
            false, a[mi].v, false, b1[nj].v, (short)0, acc1[mi][nj], false, false);
        acc3[mi][nj] = __builtin_amdgcn_wmma_f32_16x16x32_bf16(
            false, a[mi].v, false, b3[nj].v, (short)0, acc3[mi][nj], false, false);
      }
  }
  // epilogue: silu(z1) * z3 -> bf16 ; C layout: elem r => (M=r+8*half, N=l16)
#pragma unroll
  for (int mi = 0; mi < 4; ++mi)
#pragma unroll
    for (int nj = 0; nj < 2; ++nj)
#pragma unroll
      for (int r = 0; r < 8; ++r) {
        const float z = acc1[mi][nj][r];
        const float h = (z / (1.f + __expf(-z))) * acc3[mi][nj][r];
        const int m = m0 + mi * 16 + half * 8 + r;
        const int n = n0 + nj * 16 + l16;
        hout[(size_t)m * HIDDEN + n] = (__bf16)h;
      }
}

// ---------------------------------------------------------------------------
// 5) GEMM2: Y = H @ W2   (f32 out)
//    8 waves (2 x 4); block tile 128x256; wave tile 64x64
//    => 16 WMMAs per K-step vs 16 b128 loads.
// ---------------------------------------------------------------------------
__global__ __launch_bounds__(256) void gemm2(const __bf16* __restrict__ hmat,
                                             const __bf16* __restrict__ w2t,
                                             float* __restrict__ y) {
  const int mblk = blockIdx.x;
  const int e    = mblk / (CAP / 128);
  const int tid  = threadIdx.x;
  const int wave = tid >> 5, lane = tid & 31;
  const int wrow = wave & 1, wcol = wave >> 1;       // 2 x 4 wave grid
  const int half = lane >> 4, l16 = lane & 15;
  const int m0 = mblk * 128 + wrow * 64;             // 64-row wave tile
  const int n0 = blockIdx.y * 256 + wcol * 64;       // 64-col wave tile
  const __bf16* W2 = w2t + (size_t)e * D_MODEL * HIDDEN;

  v8f acc[4][4];
#pragma unroll
  for (int mi = 0; mi < 4; ++mi)
#pragma unroll
    for (int nj = 0; nj < 4; ++nj) acc[mi][nj] = (v8f)(0.f);

  for (int k0 = 0; k0 < HIDDEN; k0 += 32) {
    FragAB a[4], b[4];
#pragma unroll
    for (int mi = 0; mi < 4; ++mi)
      load_fragA(a[mi], hmat, m0 + mi * 16 + l16, k0, half);
#pragma unroll
    for (int nj = 0; nj < 4; ++nj)
      load_fragB(b[nj], W2, n0 + nj * 16 + l16, k0, half);
#pragma unroll
    for (int mi = 0; mi < 4; ++mi)
#pragma unroll
      for (int nj = 0; nj < 4; ++nj)
        acc[mi][nj] = __builtin_amdgcn_wmma_f32_16x16x32_bf16(
            false, a[mi].v, false, b[nj].v, (short)0, acc[mi][nj], false, false);
  }
#pragma unroll
  for (int mi = 0; mi < 4; ++mi)
#pragma unroll
    for (int nj = 0; nj < 4; ++nj)
#pragma unroll
      for (int r = 0; r < 8; ++r) {
        const int m = m0 + mi * 16 + half * 8 + r;
        const int n = n0 + nj * 16 + l16;
        y[(size_t)m * D_MODEL + n] = acc[mi][nj][r];
      }
}

// ---------------------------------------------------------------------------
// 6) Combine: out[tok] = sum_k gate_w * Y[slot_k]   (no atomics, full write)
// ---------------------------------------------------------------------------
__global__ __launch_bounds__(256) void combine(const float* __restrict__ y,
                                               const int* __restrict__ slot,
                                               const float* __restrict__ ew,
                                               float* __restrict__ out) {
  const int tok = blockIdx.x;
  const int c   = threadIdx.x * 8;
  const int t0  = tok * TOPK;
  const int s0 = slot[t0], s1 = slot[t0 + 1];
  const float g0 = ew[t0], g1 = ew[t0 + 1];
  float4 a0 = {0, 0, 0, 0}, a1 = a0, b0 = a0, b1 = a0;
  if (s0 >= 0) {
    const float* p = y + (size_t)s0 * D_MODEL + c;
    a0 = *(const float4*)p; a1 = *(const float4*)(p + 4);
  }
  if (s1 >= 0) {
    const float* p = y + (size_t)s1 * D_MODEL + c;
    b0 = *(const float4*)p; b1 = *(const float4*)(p + 4);
  }
  float4 o0, o1;
  o0.x = g0 * a0.x + g1 * b0.x; o0.y = g0 * a0.y + g1 * b0.y;
  o0.z = g0 * a0.z + g1 * b0.z; o0.w = g0 * a0.w + g1 * b0.w;
  o1.x = g0 * a1.x + g1 * b1.x; o1.y = g0 * a1.y + g1 * b1.y;
  o1.z = g0 * a1.z + g1 * b1.z; o1.w = g0 * a1.w + g1 * b1.w;
  float* q = out + (size_t)tok * D_MODEL + c;
  *(float4*)q = o0; *(float4*)(q + 4) = o1;
}

// ---------------------------------------------------------------------------
extern "C" void kernel_launch(void* const* d_in, const int* in_sizes, int n_in,
                              void* d_out, int out_size, void* d_ws, size_t ws_size,
                              hipStream_t stream) {
  (void)in_sizes; (void)n_in; (void)out_size; (void)ws_size;
  const float* x    = (const float*)d_in[0];
  const float* ew   = (const float*)d_in[1];
  const int*   eidx = (const int*)d_in[2];
  const float* w1   = (const float*)d_in[3];
  const float* w2   = (const float*)d_in[4];
  const float* w3   = (const float*)d_in[5];
  float* out = (float*)d_out;

  char* ws = (char*)d_ws;
  size_t off = 0;
  auto carve = [&](size_t bytes) -> char* {
    char* p = ws + off;
    off = (off + bytes + 255) & ~(size_t)255;
    return p;
  };
  const size_t WSZ = (size_t)NEXP * 2048 * 2048;    // elems per weight tensor
  int*    counts = (int*)carve(NEXP * sizeof(int));
  int*    slot   = (int*)carve(T_ASSIGN * sizeof(int));
  int*    gtok   = (int*)carve(EROWS * sizeof(int));
  __bf16* w1t    = (__bf16*)carve(WSZ * sizeof(__bf16));
  __bf16* w3t    = (__bf16*)carve(WSZ * sizeof(__bf16));
  __bf16* w2t    = (__bf16*)carve(WSZ * sizeof(__bf16));
  __bf16* xg     = (__bf16*)carve((size_t)EROWS * D_MODEL * sizeof(__bf16));
  __bf16* hbuf   = (__bf16*)carve((size_t)EROWS * HIDDEN * sizeof(__bf16));
  float*  ybuf   = (float*)carve((size_t)EROWS * D_MODEL * sizeof(float));

  int* counts_out = (int*)(out + (size_t)N_TOK * D_MODEL);  // tuple tail

  moe_route<<<1, 256, 0, stream>>>(eidx, counts, slot, gtok, counts_out);

  dim3 tb(32, 32), tg(64, 64, NEXP);
  transpose_cvt<<<tg, tb, 0, stream>>>(w1, w1t);
  transpose_cvt<<<tg, tb, 0, stream>>>(w3, w3t);
  transpose_cvt<<<tg, tb, 0, stream>>>(w2, w2t);

  gather_xg<<<EROWS, 256, 0, stream>>>(x, counts, gtok, xg);

  dim3 g1(EROWS / 128, HIDDEN / 128);
  gemm1_swiglu<<<g1, 256, 0, stream>>>(xg, w1t, w3t, hbuf);

  dim3 g2(EROWS / 128, D_MODEL / 256);
  gemm2<<<g2, 256, 0, stream>>>(hbuf, w2t, ybuf);

  combine<<<N_TOK, 256, 0, stream>>>(ybuf, slot, ew, out);
}
